// modeler_15882789060866
// MI455X (gfx1250) — compile-verified, matrix operand order checked by reading, not run on hardware
//
#include <hip/hip_runtime.h>

typedef __attribute__((ext_vector_type(16))) __bf16 v16bf;
typedef __attribute__((ext_vector_type(8)))  float  v8f;

__device__ __forceinline__ __bf16 f2bf(float x) { return (__bf16)x; }

// ---------------------------------------------------------------------------
// Weight repack: W[K,N] f32 row-major -> bf16 in exact WMMA B-fragment order:
//   P[(kb/32) * (N/16) + ntile][lane][j]  (16 bf16 = 32 B per lane)
// Lane layout (ISA 7.12.2, 16-bit B): lane = half*16 + l15, column = ntile*16
// + l15; j=0..7 -> K = kb + half*8 + j, j=8..15 -> K = kb + 16 + half*8 + j-8.
// One thread emits one lane's 16 bf16 values.
// ---------------------------------------------------------------------------
__global__ void pack_w(const float* __restrict__ W, __bf16* __restrict__ P,
                       int K, int N)
{
    const int t = blockIdx.x * blockDim.x + threadIdx.x;
    const int nt = N >> 4;
    const int total = (K >> 5) * nt * 32;
    if (t >= total) return;
    const int lane = t & 31;
    const int tile = t >> 5;                 // kbi * nt + ntile
    const int kb   = (tile / nt) << 5;
    const int ntile = tile - (tile / nt) * nt;
    const int half = lane >> 4;
    const int l15  = lane & 15;
    const int col  = ntile * 16 + l15;
    const int k0   = kb + half * 8;
    __bf16* dst = P + ((size_t)t << 4);
#pragma unroll
    for (int j = 0; j < 8; ++j) {
        dst[j]     = f2bf(W[(size_t)(k0 + j) * N + col]);
        dst[j + 8] = f2bf(W[(size_t)(k0 + 16 + j) * N + col]);
    }
}

// ---------------------------------------------------------------------------
// WMMA GEMM, fused bias + optional PReLU. One wave computes a 16x64 strip of
// C (4 accumulator tiles sharing one A fragment -> 4 WMMAs per K-step).
//   C[M,N] = act( A_cat[M,K] @ W[K,N] + bias )
// A_cat row r: cols [0,ksplit) from A (lda), cols [ksplit,K) from A2 (lda2);
// ksplit % 32 == 0 so each K-step reads one side only. Wp = packed bf16
// weights from pack_w. Requires N % 64 == 0.
// ---------------------------------------------------------------------------
__global__ __launch_bounds__(256)
void wmma_gemm_bias_act(const float* __restrict__ A, int lda,
                        const float* __restrict__ A2, int lda2, int ksplit,
                        const __bf16* __restrict__ Wp,
                        const float* __restrict__ bias,
                        const float* __restrict__ slope,   // nullptr => identity
                        float* __restrict__ C, int ldc,
                        int M, int K, int N)
{
    const int lane   = threadIdx.x & 31;
    const int wid    = (blockIdx.x * blockDim.x + threadIdx.x) >> 5;  // wave-uniform
    const int ngrp   = N >> 6;                 // 64-column groups
    const int mtiles = (M + 15) >> 4;
    if (wid >= mtiles * ngrp) return;          // whole wave exits together
    const int mtile  = wid / ngrp;
    const int ngroup = wid - mtile * ngrp;
    const int ntile0 = ngroup << 2;
    const int nt     = N >> 4;

    const int half = lane >> 4;
    const int l15  = lane & 15;
    int arow = mtile * 16 + l15;
    if (arow >= M) arow = M - 1;               // clamp; stores guarded below

    v8f acc0 = {}, acc1 = {}, acc2 = {}, acc3 = {};
    const int kblocks = K >> 5;
    for (int kbi = 0; kbi < kblocks; ++kbi) {
        const int kb = kbi << 5;
        const float* arp = (A2 && kb >= ksplit)
                               ? A2 + (size_t)arow * lda2 + (kb - ksplit)
                               : A  + (size_t)arow * lda  + kb;
        __builtin_prefetch(arp + 32, 0, 1);    // next K-step of the A stream
        const int k0 = half * 8;
        v16bf a;
#pragma unroll
        for (int j = 0; j < 8; ++j) {
            a[j]     = f2bf(arp[k0 + j]);
            a[j + 8] = f2bf(arp[k0 + 16 + j]);
        }
        // Packed B: consecutive ntiles are 32*16 bf16 apart.
        const __bf16* wb = Wp + ((((size_t)kbi * nt + ntile0) * 32 + lane) << 4);
        const v16bf b0 = *(const v16bf*)(wb);
        const v16bf b1 = *(const v16bf*)(wb + 512);
        const v16bf b2 = *(const v16bf*)(wb + 1024);
        const v16bf b3 = *(const v16bf*)(wb + 1536);
        acc0 = __builtin_amdgcn_wmma_f32_16x16x32_bf16(false, a, false, b0, (short)0, acc0, false, false);
        acc1 = __builtin_amdgcn_wmma_f32_16x16x32_bf16(false, a, false, b1, (short)0, acc1, false, false);
        acc2 = __builtin_amdgcn_wmma_f32_16x16x32_bf16(false, a, false, b2, (short)0, acc2, false, false);
        acc3 = __builtin_amdgcn_wmma_f32_16x16x32_bf16(false, a, false, b3, (short)0, acc3, false, false);
    }

    const float sl = slope ? slope[0] : 0.0f;
    const int rbase = mtile * 16 + half * 8;   // C layout: vgpr r -> M = rbase + r
    const v8f accs[4] = {acc0, acc1, acc2, acc3};
#pragma unroll
    for (int t = 0; t < 4; ++t) {
        const int ccol = (ntile0 + t) * 16 + l15;
        const float bv = bias[ccol];
#pragma unroll
        for (int r = 0; r < 8; ++r) {
            const int row = rbase + r;
            if (row < M) {
                float v = accs[t][r] + bv;
                if (slope) v = (v >= 0.0f) ? v : sl * v;
                C[(size_t)row * ldc + ccol] = v;
            }
        }
    }
}

// ---------------------------------------------------------------------------
// Per-edge degree counts (shared by all three mean stages).
// ---------------------------------------------------------------------------
__global__ void edge_deg(const int* __restrict__ ev, const int* __restrict__ eu,
                         float* __restrict__ degv, float* __restrict__ degu, int ne)
{
    const int e = blockIdx.x * blockDim.x + threadIdx.x;
    if (e >= ne) return;
    atomicAdd(&degv[ev[e]], 1.0f);
    atomicAdd(&degu[eu[e]], 1.0f);
}

// ---------------------------------------------------------------------------
// Bidirectional edge scatter-add:
//   accv[ev[e]] += fu[eu[e]]   and   accu[eu[e]] += fv[ev[e]]
// One thread handles a 4-float chunk of one edge (accumulators are L2-resident:
// 50000 x 256 f32 = 51 MB << 192 MB L2, so atomics resolve at L2 bandwidth).
// ---------------------------------------------------------------------------
__global__ void edge_scatter(const int* __restrict__ eu, const int* __restrict__ ev,
                             const float* __restrict__ fu, float* __restrict__ accv,
                             const float* __restrict__ fv, float* __restrict__ accu,
                             int ne, int width)
{
    const int chunks = width >> 2;
    const int t = blockIdx.x * blockDim.x + threadIdx.x;
    if (t >= ne * chunks) return;
    const int e = t / chunks;
    const int c = (t - e * chunks) << 2;
    const int u = eu[e], v = ev[e];
    const float* su = fu + (size_t)u * width + c;
    const float* sv = fv + (size_t)v * width + c;
    float* dv = accv + (size_t)v * width + c;
    float* du = accu + (size_t)u * width + c;
#pragma unroll
    for (int j = 0; j < 4; ++j) {
        atomicAdd(&dv[j], su[j]);
        atomicAdd(&du[j], sv[j]);
    }
}

// buf[row][:] /= max(deg[row], 1)
__global__ void div_deg(float* __restrict__ buf, const float* __restrict__ deg,
                        int n, int width)
{
    const int i = blockIdx.x * blockDim.x + threadIdx.x;
    if (i >= n * width) return;
    const int row = i / width;
    buf[i] /= fmaxf(deg[row], 1.0f);
}

// out[row][colofs + col] = (sum[row][col] + e3[row][col]) / (deg[row] + 1)
__global__ void final_combine(const float* __restrict__ sum, const float* __restrict__ e3,
                              const float* __restrict__ deg,
                              float* __restrict__ out, int n, int width, int ldo, int colofs)
{
    const int i = blockIdx.x * blockDim.x + threadIdx.x;
    if (i >= n * width) return;
    const int row = i / width;
    const int col = i - row * width;
    out[(size_t)row * ldo + colofs + col] = (sum[i] + e3[i]) / (deg[row] + 1.0f);
}

// ---------------------------------------------------------------------------
extern "C" void kernel_launch(void* const* d_in, const int* in_sizes, int n_in,
                              void* d_out, int out_size, void* d_ws, size_t ws_size,
                              hipStream_t stream)
{
    const float* features_v = (const float*)d_in[0];
    const float* features_u = (const float*)d_in[1];
    const int*   edge_v     = (const int*)d_in[2];
    const int*   edge_u     = (const int*)d_in[3];
    const float* Wv1 = (const float*)d_in[4];
    const float* bv1 = (const float*)d_in[5];
    const float* av1 = (const float*)d_in[6];
    const float* Wu1 = (const float*)d_in[7];
    const float* bu1 = (const float*)d_in[8];
    const float* au1 = (const float*)d_in[9];
    const float* Wv2 = (const float*)d_in[10];
    const float* bv2 = (const float*)d_in[11];
    const float* av2 = (const float*)d_in[12];
    const float* Wu2 = (const float*)d_in[13];
    const float* bu2 = (const float*)d_in[14];
    const float* au2 = (const float*)d_in[15];
    const float* Wv3 = (const float*)d_in[16];
    const float* bv3 = (const float*)d_in[17];
    const float* Wu3 = (const float*)d_in[18];
    const float* bu3 = (const float*)d_in[19];

    const int HID = in_sizes[5];            // 256
    const int FT  = in_sizes[4] / HID;      // 128
    const int OUT = in_sizes[11];           // 128
    const int NV  = in_sizes[0] / FT;       // 50000
    const int NU  = in_sizes[1] / FT;       // 50000
    const int NE  = in_sizes[2];            // 800000
    const int LDO = 2 * OUT;                // 256 (output row stride)
    const int K3  = OUT + FT;               // 256 (stage-3 concat K)

    // Workspace layout (floats), with reuse:
    //   degv[NV] degu[NU] | m1v[NV*FT] m1u[NU*FT] | ve1[NV*HID] ue1[NU*HID]
    //   a2v[NV*HID] a2u[NU*HID] | packed bf16 weights (64B-aligned)
    //   m1v/m1u reused as ve3/ue3 (width OUT); a2v/a2u reused as sumV/sumU.
    float* ws   = (float*)d_ws;
    float* degv = ws;
    float* degu = degv + NV;
    float* m1v  = degu + NU;
    float* m1u  = m1v + (size_t)NV * FT;
    float* ve1  = m1u + (size_t)NU * FT;
    float* ue1  = ve1 + (size_t)NV * HID;
    float* a2v  = ue1 + (size_t)NU * HID;
    float* a2u  = a2v + (size_t)NV * HID;

    size_t packOff = (size_t)(a2u + (size_t)NU * HID - ws);
    packOff = (packOff + 15) & ~(size_t)15;        // 64-byte align
    __bf16* pWv1 = (__bf16*)(ws + packOff);
    __bf16* pWu1 = pWv1 + (size_t)FT * HID;
    __bf16* pWv2 = pWu1 + (size_t)FT * HID;
    __bf16* pWu2 = pWv2 + (size_t)HID * OUT;
    __bf16* pWv3 = pWu2 + (size_t)HID * OUT;
    __bf16* pWu3 = pWv3 + (size_t)K3 * OUT;

    float* out_v = (float*)d_out;                  // [NV, 256]: ve2 | sv
    float* out_u = out_v + (size_t)NV * LDO;       // [NU, 256]: ue2 | su

    auto gemm_blocks = [](int M, int N) {
        const int waves = ((M + 15) / 16) * (N / 64);
        return (waves * 32 + 255) / 256;
    };
    auto pack_blocks = [](int K, int N) {
        const int total = (K / 32) * (N / 16) * 32;
        return (total + 255) / 256;
    };
    const int T = 256;

    // --- repack static weights into bf16 WMMA fragment order (L2-resident) ---
    pack_w<<<pack_blocks(FT, HID), T, 0, stream>>>(Wv1, pWv1, FT, HID);
    pack_w<<<pack_blocks(FT, HID), T, 0, stream>>>(Wu1, pWu1, FT, HID);
    pack_w<<<pack_blocks(HID, OUT), T, 0, stream>>>(Wv2, pWv2, HID, OUT);
    pack_w<<<pack_blocks(HID, OUT), T, 0, stream>>>(Wu2, pWu2, HID, OUT);
    pack_w<<<pack_blocks(K3, OUT), T, 0, stream>>>(Wv3, pWv3, K3, OUT);
    pack_w<<<pack_blocks(K3, OUT), T, 0, stream>>>(Wu3, pWu3, K3, OUT);

    // --- stage 0: degrees + stage-1 accumulators (one contiguous memset) ---
    hipMemsetAsync(degv, 0,
                   sizeof(float) * ((size_t)NV + NU + (size_t)NV * FT + (size_t)NU * FT),
                   stream);
    edge_deg<<<(NE + T - 1) / T, T, 0, stream>>>(edge_v, edge_u, degv, degu, NE);

    // --- stage 1: mean of raw neighbor features, then GEMM+PReLU ---
    {
        const int tot = NE * (FT / 4);
        edge_scatter<<<(tot + T - 1) / T, T, 0, stream>>>(
            edge_u, edge_v, features_u, m1v, features_v, m1u, NE, FT);
    }
    div_deg<<<((NV * FT) + T - 1) / T, T, 0, stream>>>(m1v, degv, NV, FT);
    div_deg<<<((NU * FT) + T - 1) / T, T, 0, stream>>>(m1u, degu, NU, FT);
    wmma_gemm_bias_act<<<gemm_blocks(NV, HID), T, 0, stream>>>(
        m1v, FT, nullptr, 0, FT, pWv1, bv1, av1, ve1, HID, NV, FT, HID);
    wmma_gemm_bias_act<<<gemm_blocks(NU, HID), T, 0, stream>>>(
        m1u, FT, nullptr, 0, FT, pWu1, bu1, au1, ue1, HID, NU, FT, HID);

    // --- stage 2: mean of transformed neighbors, GEMM+PReLU -> d_out[:,0:OUT]
    hipMemsetAsync(a2v, 0, sizeof(float) * ((size_t)NV + NU) * HID, stream);
    {
        const int tot = NE * (HID / 4);
        edge_scatter<<<(tot + T - 1) / T, T, 0, stream>>>(
            edge_u, edge_v, ue1, a2v, ve1, a2u, NE, HID);
    }
    div_deg<<<((NV * HID) + T - 1) / T, T, 0, stream>>>(a2v, degv, NV, HID);
    div_deg<<<((NU * HID) + T - 1) / T, T, 0, stream>>>(a2u, degu, NU, HID);
    wmma_gemm_bias_act<<<gemm_blocks(NV, OUT), T, 0, stream>>>(
        a2v, HID, nullptr, 0, HID, pWv2, bv2, av2, out_v, LDO, NV, HID, OUT);
    wmma_gemm_bias_act<<<gemm_blocks(NU, OUT), T, 0, stream>>>(
        a2u, HID, nullptr, 0, HID, pWu2, bu2, au2, out_u, LDO, NU, HID, OUT);

    // --- stage 3: concat([e2, features]) GEMM (identity act) -> e3 ---
    wmma_gemm_bias_act<<<gemm_blocks(NV, OUT), T, 0, stream>>>(
        out_v, LDO, features_v, FT, OUT, pWv3, bv3, nullptr, m1v /*ve3*/, OUT,
        NV, K3, OUT);
    wmma_gemm_bias_act<<<gemm_blocks(NU, OUT), T, 0, stream>>>(
        out_u, LDO, features_u, FT, OUT, pWu3, bu3, nullptr, m1u /*ue3*/, OUT,
        NU, K3, OUT);

    // --- final: self-inclusive neighbor mean -> d_out[:,OUT:2*OUT] ---
    hipMemsetAsync(a2v, 0, sizeof(float) * (size_t)NV * OUT, stream);   // sumV
    hipMemsetAsync(a2u, 0, sizeof(float) * (size_t)NU * OUT, stream);   // sumU
    {
        const int tot = NE * (OUT / 4);
        edge_scatter<<<(tot + T - 1) / T, T, 0, stream>>>(
            edge_u, edge_v, m1u /*ue3*/, a2v /*sumV*/, m1v /*ve3*/, a2u /*sumU*/,
            NE, OUT);
    }
    final_combine<<<((NV * OUT) + T - 1) / T, T, 0, stream>>>(
        a2v, m1v, degv, out_v, NV, OUT, LDO, OUT);
    final_combine<<<((NU * OUT) + T - 1) / T, T, 0, stream>>>(
        a2u, m1u, degu, out_u, NU, OUT, LDO, OUT);
}
